// MoE_46325517255234
// MI455X (gfx1250) — matrix operand order repaired and checked by reading, not compiled, not run on hardware
//
#include <hip/hip_runtime.h>
#include <hip/hip_bf16.h>

typedef __attribute__((ext_vector_type(16))) _Float16 v16h;
typedef __attribute__((ext_vector_type(4)))  _Float16 v4h;
typedef __attribute__((ext_vector_type(8)))  float    v8f;
typedef __attribute__((ext_vector_type(4)))  float    v4f;

#define B_TOK 2048
#define S_SEQ 256
#define H_DIM 1024
#define N_EXP 8

// ---------------------------------------------------------------------------
// Kernel 1: mean over S  (bandwidth bound: 2.1 GB read)  + fp16 copy of x
// one thread per float4 of [B,H]
// ---------------------------------------------------------------------------
__global__ void moe_mean_kernel(const v4f* __restrict__ hs,
                                v4f* __restrict__ x4,
                                v4h* __restrict__ xh4) {
    int idx = blockIdx.x * blockDim.x + threadIdx.x;   // B * H/4 threads
    int b  = idx >> 8;                                  // H/4 = 256
    int hv = idx & 255;
    const v4f* p = hs + (size_t)b * S_SEQ * 256 + hv;
    v4f acc = {0.f, 0.f, 0.f, 0.f};
#pragma unroll 4
    for (int s = 0; s < S_SEQ; ++s) {
        v4f v = __builtin_nontemporal_load(p + (size_t)s * 256);
        acc += v;
    }
    const float inv = 1.0f / (float)S_SEQ;
    v4f r = acc * inv;
    x4[idx] = r;
    v4h h;
    h.x = (_Float16)r.x; h.y = (_Float16)r.y;
    h.z = (_Float16)r.z; h.w = (_Float16)r.w;
    xh4[idx] = h;
}

// ---------------------------------------------------------------------------
// Kernel 2: transpose + fp16-convert expert weights:  whT[e][n][k] = w[e][k][n]
// block (32,8), grid (H/32, H/32, E)
// ---------------------------------------------------------------------------
__global__ void moe_wt_kernel(const float* __restrict__ w,
                              _Float16* __restrict__ whT) {
    __shared__ _Float16 t[32][33];
    int e = blockIdx.z;
    int n0 = blockIdx.x * 32, k0 = blockIdx.y * 32;
    int tx = threadIdx.x, ty = threadIdx.y;
    const float* wp = w + (size_t)e * H_DIM * H_DIM;
#pragma unroll
    for (int r = 0; r < 4; ++r) {
        int k = k0 + ty + r * 8;
        t[tx][ty + r * 8] = (_Float16)wp[(size_t)k * H_DIM + n0 + tx];
    }
    __syncthreads();
    _Float16* op = whT + (size_t)e * H_DIM * H_DIM;
#pragma unroll
    for (int r = 0; r < 4; ++r) {
        int n = n0 + ty + r * 8;
        op[(size_t)n * H_DIM + k0 + tx] = t[ty + r * 8][tx];
    }
}

// ---------------------------------------------------------------------------
// Kernel 3a: zero the per-expert token counters
// ---------------------------------------------------------------------------
__global__ void moe_zero_kernel(unsigned int* counts) {
    if (threadIdx.x < N_EXP) counts[threadIdx.x] = 0u;
}

// ---------------------------------------------------------------------------
// Kernel 3b: gating scores (fp32) + top-2 + build per-expert token lists
// one wave per token; lane = (chunk<<3)|expert
// ---------------------------------------------------------------------------
__global__ void moe_gate_kernel(const float* __restrict__ x,
                                const float* __restrict__ gw,
                                const float* __restrict__ gb,
                                unsigned int* __restrict__ counts,
                                unsigned int* __restrict__ ent_tok,
                                float* __restrict__ ent_w) {
    int wave = threadIdx.x >> 5;
    int lane = threadIdx.x & 31;
    int b = blockIdx.x * 8 + wave;
    int e = lane & 7;
    int c = lane >> 3;                       // 4 chunks of 256
    const float* xp = x + (size_t)b * H_DIM + c * 256;
    const float* gp = gw + (size_t)c * 256 * N_EXP + e;
    float p = 0.f;
#pragma unroll 8
    for (int i = 0; i < 256; ++i) p += xp[i] * gp[i * N_EXP];
    p += __shfl_xor(p, 8, 32);
    p += __shfl_xor(p, 16, 32);
    float score = p + gb[e];
    float s[N_EXP];
#pragma unroll
    for (int j = 0; j < N_EXP; ++j) s[j] = __shfl(score, j, 32);
    if (lane == 0) {
        int e0 = 0;
        for (int j = 1; j < N_EXP; ++j) if (s[j] > s[e0]) e0 = j;
        int e1 = (e0 == 0) ? 1 : 0;
        for (int j = 0; j < N_EXP; ++j) if (j != e0 && s[j] > s[e1]) e1 = j;
        unsigned int p0 = atomicAdd(&counts[e0], 1u);
        ent_tok[e0 * B_TOK + p0] = (unsigned int)(b * 2 + 0);
        ent_w  [e0 * B_TOK + p0] = s[e0];
        unsigned int p1 = atomicAdd(&counts[e1], 1u);
        ent_tok[e1 * B_TOK + p1] = (unsigned int)(b * 2 + 1);
        ent_w  [e1 * B_TOK + p1] = s[e1];
    }
}

// ---------------------------------------------------------------------------
// Kernel 4: gathered per-expert GEMM with WMMA f32_16x16x32_f16
// grid.x = E * (B/64) token tiles, grid.y = H/128 column tiles
// block = 256 threads = 8 waves, wave = (rowblock 0..3) x (nhalf 0..1)
// each wave: 16 rows x 64 cols = 4 x v8f accumulators
// A frag: lane<16 -> M=lane, K in {k0+hi*8 .. +7} and {k0+16+hi*8 .. +7}
// B frag: lane<16 -> N=lane, K in {k0+hi*8 .. +7} and {k0+16+hi*8 .. +7} (whT[n][k])
// ---------------------------------------------------------------------------
__global__ void moe_gemm_kernel(const _Float16* __restrict__ xh,
                                const _Float16* __restrict__ whT,
                                const float* __restrict__ eb,
                                const unsigned int* __restrict__ ent_tok,
                                const float* __restrict__ ent_w,
                                const unsigned int* __restrict__ counts,
                                float* __restrict__ y) {
    const int e    = blockIdx.x >> 5;       // 32 tiles per expert
    const int tile = blockIdx.x & 31;
    const unsigned int cnt = counts[e];
    const unsigned int m0  = tile * 64u;
    if (m0 >= cnt) return;

    __shared__ unsigned int s_tok[64];
    __shared__ float        s_w[64];
    int tid = threadIdx.x;
    if (tid < 64) {
        unsigned int idx = m0 + (unsigned int)tid;
        s_tok[tid] = (idx < cnt) ? ent_tok[e * B_TOK + idx] : 0u;
        s_w[tid]   = (idx < cnt) ? ent_w  [e * B_TOK + idx] : 0.0f;
    }
    __syncthreads();

    const int wave = tid >> 5, lane = tid & 31;
    const int rm = wave >> 1;               // row block 0..3
    const int wn = wave & 1;                // n half 0..1
    const int hi = lane >> 4;               // upper half-wave
    const int ln = lane & 15;
    const int nbase = blockIdx.y * 128 + wn * 64;

    const int arow = rm * 16 + ln;
    const _Float16* aptr = xh + (size_t)(s_tok[arow] >> 1) * H_DIM;
    const _Float16* bbase = whT + ((size_t)e * H_DIM + nbase + ln) * H_DIM;

    v8f acc[4] = {};
#pragma unroll 2
    for (int k0 = 0; k0 < H_DIM; k0 += 32) {
        v16h a;
        const _Float16* ap = aptr + k0 + hi * 8;
        *((uint4*)&a)     = *((const uint4*)ap);          // K group 0..7
        *((uint4*)&a + 1) = *((const uint4*)(ap + 16));   // K group 16..23
#pragma unroll
        for (int nt = 0; nt < 4; ++nt) {
            v16h bf;
            const _Float16* bp = bbase + (size_t)nt * 16 * H_DIM + k0 + hi * 8;
            *((uint4*)&bf)     = *((const uint4*)bp);
            *((uint4*)&bf + 1) = *((const uint4*)(bp + 16));
            acc[nt] = __builtin_amdgcn_wmma_f32_16x16x32_f16(
                false, a, false, bf, (short)0, acc[nt], false, false);
        }
    }

#pragma unroll
    for (int nt = 0; nt < 4; ++nt) {
        int col = nbase + nt * 16 + ln;
        float bias = eb[e * H_DIM + col];
#pragma unroll
        for (int r = 0; r < 8; ++r) {
            int row = rm * 16 + hi * 8 + r;          // D: M = r + hi*8
            unsigned int idx = m0 + (unsigned int)row;
            if (idx < cnt) {
                unsigned int ts = s_tok[row];         // tok*2 + slot
                y[(size_t)ts * H_DIM + col] = s_w[row] * (acc[nt][r] + bias);
            }
        }
    }
}

// ---------------------------------------------------------------------------
// Kernel 5: combine the two slots per token
// ---------------------------------------------------------------------------
__global__ void moe_combine_kernel(const v4f* __restrict__ y4,
                                   v4f* __restrict__ out4) {
    int idx = blockIdx.x * blockDim.x + threadIdx.x;   // B * H/4
    int b  = idx >> 8;
    int hv = idx & 255;
    v4f a = y4[(size_t)(2 * b)     * 256 + hv];
    v4f c = y4[(size_t)(2 * b + 1) * 256 + hv];
    out4[idx] = a + c;
}

// ---------------------------------------------------------------------------
extern "C" void kernel_launch(void* const* d_in, const int* in_sizes, int n_in,
                              void* d_out, int out_size, void* d_ws, size_t ws_size,
                              hipStream_t stream) {
    const float* hs = (const float*)d_in[0];   // [B,S,H]
    const float* gw = (const float*)d_in[1];   // [H,E]
    const float* gb = (const float*)d_in[2];   // [E]
    const float* ew = (const float*)d_in[3];   // [E,H,H]
    const float* eb = (const float*)d_in[4];   // [E,H]
    float* out = (float*)d_out;                // [B,H]

    char* ws = (char*)d_ws;
    const size_t MB = 1024u * 1024u;
    float*        x      = (float*)(ws);                    //  8 MB [B,H] f32
    _Float16*     xh     = (_Float16*)(ws +  8 * MB);       //  4 MB [B,H] f16
    _Float16*     whT    = (_Float16*)(ws + 12 * MB);       // 16 MB [E,H,H] f16 (n-major)
    float*        y      = (float*)(ws + 28 * MB);          // 16 MB [B,2,H] f32
    unsigned int* etok   = (unsigned int*)(ws + 44 * MB);   // 64 KB
    float*        ewgt   = (float*)(ws + 44 * MB + 65536);  // 64 KB
    unsigned int* counts = (unsigned int*)(ws + 44 * MB + 2 * 65536);

    // 1. mean over S (+ fp16 copy)
    moe_mean_kernel<<<(B_TOK * (H_DIM / 4)) / 256, 256, 0, stream>>>(
        (const v4f*)hs, (v4f*)x, (v4h*)xh);

    // 2. fp16 transposed expert weights
    moe_wt_kernel<<<dim3(H_DIM / 32, H_DIM / 32, N_EXP), dim3(32, 8), 0, stream>>>(
        ew, whT);

    // 3. gating + top-2 routing lists
    moe_zero_kernel<<<1, 32, 0, stream>>>(counts);
    moe_gate_kernel<<<B_TOK / 8, 256, 0, stream>>>(x, gw, gb, counts, etok, ewgt);

    // 4. per-expert WMMA GEMM into [B,2,H] slots (weight + bias applied)
    moe_gemm_kernel<<<dim3(N_EXP * (B_TOK / 64), H_DIM / 128), 256, 0, stream>>>(
        xh, whT, eb, etok, ewgt, counts, y);

    // 5. combine slots
    moe_combine_kernel<<<(B_TOK * (H_DIM / 4)) / 256, 256, 0, stream>>>(
        (const v4f*)y, (v4f*)out);
}
